// Model_37692632990204
// MI455X (gfx1250) — matrix operand order repaired and checked by reading, not compiled, not run on hardware
//
#include <hip/hip_runtime.h>
#include <math.h>

typedef __attribute__((ext_vector_type(16))) _Float16 v16h;
typedef __attribute__((ext_vector_type(8)))  _Float16 v8h;
typedef __attribute__((ext_vector_type(8)))  float    v8f;
typedef __attribute__((ext_vector_type(4)))  float    v4f;

#define WMMA16(af, bf, cc) \
  __builtin_amdgcn_wmma_f32_16x16x32_f16(false, (af), false, (bf), (short)0, (cc), false, false)

// ---------------------------------------------------------------------------
// Fragment helpers (wave32).
// A-frag (16x32 f16, ISA 7.12.2): lane L(<16): row M=L, K = {k0..k0+7, k0+16..k0+23}
//                                 lane L+16 : row M=L, K = {k0+8..k0+15, k0+24..k0+31}
// B-frag (32x16 f16): lane L(<16): col N=L, K = {k0..k0+7, k0+16..k0+23}; lane L+16: {k0+8..k0+15, k0+24..k0+31}
// C/D  (16x16 f32)  : lane L(<16): (M=r, N=L) in reg r; lane L+16: (M=r+8, N=L-16)
// ---------------------------------------------------------------------------

// streamed fp32 A (one-shot, 307MB > L2): non-temporal loads
__device__ inline v8h cvt8_f32_nt(const float* __restrict__ p) {
  const v4f x = __builtin_nontemporal_load((const v4f*)p);
  const v4f y = __builtin_nontemporal_load(((const v4f*)p) + 1);
  v8h r;
  r[0] = (_Float16)x[0]; r[1] = (_Float16)x[1]; r[2] = (_Float16)x[2]; r[3] = (_Float16)x[3];
  r[4] = (_Float16)y[0]; r[5] = (_Float16)y[1]; r[6] = (_Float16)y[2]; r[7] = (_Float16)y[3];
  return r;
}

__device__ inline v16h load_bfrag(const _Float16* __restrict__ BT, size_t nrow, int Kd,
                                  int k0, int hf) {
  const v8h* p = (const v8h*)(BT + nrow * (size_t)Kd + k0 + hf * 8);
  v16h b;
  ((v8h*)&b)[0] = p[0];
  ((v8h*)&b)[1] = p[2];
  return b;
}

// streamed f16 A (H1 activations, one-shot): non-temporal loads
__device__ inline v16h load_afrag_f16_nt(const _Float16* __restrict__ A, size_t row, int Kd,
                                         int ka) {
  const _Float16* base = A + row * (size_t)Kd + ka;
  v16h a;
  ((v8h*)&a)[0] = __builtin_nontemporal_load((const v8h*)(base));
  ((v8h*)&a)[1] = __builtin_nontemporal_load((const v8h*)(base + 16));
  return a;
}

// pair_emb row chunk: k<768 -> me[m], k<1536 -> me[j], else me[m]*me[j]
// (mention table is 2.3MB, L2/L0-hot: regular temporal loads)
__device__ inline v8h pair_chunk(const _Float16* __restrict__ me, int m, int j, int kc) {
  if (kc < 768) {
    return *(const v8h*)(me + (size_t)m * 768 + kc);
  } else if (kc < 1536) {
    return *(const v8h*)(me + (size_t)j * 768 + (kc - 768));
  } else {
    const v8h x = *(const v8h*)(me + (size_t)m * 768 + (kc - 1536));
    const v8h y = *(const v8h*)(me + (size_t)j * 768 + (kc - 1536));
    v8h r;
#pragma unroll
    for (int q = 0; q < 8; ++q) r[q] = (_Float16)((float)x[q] * (float)y[q]);
    return r;
  }
}

// ---------------------------------------------------------------------------
// GEMM 1: C_f16[M,N] = relu(A_f32[M,Kd] @ BT^T + bias), BT is f16 [N][Kd]
// Wave tile 32x64 (8 accum), block = 2(M) x 4(N) waves = 64 x 256.
// ---------------------------------------------------------------------------
__global__ __launch_bounds__(256) void gemm_a32_relu_h16(
    const float* __restrict__ A, int M, int Kd,
    const _Float16* __restrict__ BT, const float* __restrict__ bias,
    _Float16* __restrict__ C, int N) {
  const int wid = threadIdx.x >> 5, lane = threadIdx.x & 31;
  const int hf = lane >> 4, l16 = lane & 15;
  const int wm = wid & 1, wn = wid >> 1;
  const int m0 = blockIdx.x * 64 + wm * 32;
  const int n0 = blockIdx.y * 256 + wn * 64;
  int ar0 = m0 + l16;      if (ar0 >= M) ar0 = M - 1;
  int ar1 = m0 + 16 + l16; if (ar1 >= M) ar1 = M - 1;
  const float* Arow0 = A + (size_t)ar0 * Kd;
  const float* Arow1 = A + (size_t)ar1 * Kd;

  v8f acc[2][4] = {};
  for (int k0 = 0; k0 < Kd; k0 += 32) {
    const int ka = k0 + hf * 8;
    v16h af0, af1;
    ((v8h*)&af0)[0] = cvt8_f32_nt(Arow0 + ka);
    ((v8h*)&af0)[1] = cvt8_f32_nt(Arow0 + ka + 16);
    ((v8h*)&af1)[0] = cvt8_f32_nt(Arow1 + ka);
    ((v8h*)&af1)[1] = cvt8_f32_nt(Arow1 + ka + 16);
#pragma unroll
    for (int t = 0; t < 4; ++t) {
      v16h bf = load_bfrag(BT, (size_t)(n0 + t * 16 + l16), Kd, k0, hf);
      acc[0][t] = WMMA16(af0, bf, acc[0][t]);
      acc[1][t] = WMMA16(af1, bf, acc[1][t]);
    }
  }
#pragma unroll
  for (int s = 0; s < 2; ++s) {
#pragma unroll
    for (int t = 0; t < 4; ++t) {
      const int n = n0 + t * 16 + l16;
      const float bv = bias[n];
#pragma unroll
      for (int r = 0; r < 8; ++r) {
        const int m = m0 + s * 16 + hf * 8 + r;
        if (m < M) {
          float v = acc[s][t][r] + bv;
          v = v > 0.f ? v : 0.f;
          __builtin_nontemporal_store((_Float16)v, &C[(size_t)m * N + n]);
        }
      }
    }
  }
}

// ---------------------------------------------------------------------------
// GEMM 2 (fused): part[slice][m] = sum_n relu(A_f16@BT^T + bias)[m,n] * w[n]
// over this wave's 64-column slice; slices deterministic (no atomics).
// ---------------------------------------------------------------------------
__global__ __launch_bounds__(256) void gemm_a16_relu_dot(
    const _Float16* __restrict__ A, int M, int Kd,
    const _Float16* __restrict__ BT, const float* __restrict__ bias,
    const float* __restrict__ w, float* __restrict__ part, int partStride, int N) {
  const int wid = threadIdx.x >> 5, lane = threadIdx.x & 31;
  const int hf = lane >> 4, l16 = lane & 15;
  const int wm = wid & 1, wn = wid >> 1;
  const int m0 = blockIdx.x * 64 + wm * 32;
  const int n0 = blockIdx.y * 256 + wn * 64;
  int ar0 = m0 + l16;      if (ar0 >= M) ar0 = M - 1;
  int ar1 = m0 + 16 + l16; if (ar1 >= M) ar1 = M - 1;

  v8f acc[2][4] = {};
  for (int k0 = 0; k0 < Kd; k0 += 32) {
    const int ka = k0 + hf * 8;
    v16h af0 = load_afrag_f16_nt(A, (size_t)ar0, Kd, ka);
    v16h af1 = load_afrag_f16_nt(A, (size_t)ar1, Kd, ka);
#pragma unroll
    for (int t = 0; t < 4; ++t) {
      v16h bf = load_bfrag(BT, (size_t)(n0 + t * 16 + l16), Kd, k0, hf);
      acc[0][t] = WMMA16(af0, bf, acc[0][t]);
      acc[1][t] = WMMA16(af1, bf, acc[1][t]);
    }
  }
  float rs[2][8] = {};
#pragma unroll
  for (int t = 0; t < 4; ++t) {
    const int n = n0 + t * 16 + l16;
    const float bv = bias[n], wv = w[n];
#pragma unroll
    for (int s = 0; s < 2; ++s) {
#pragma unroll
      for (int r = 0; r < 8; ++r) {
        float v = acc[s][t][r] + bv;
        v = v > 0.f ? v : 0.f;
        rs[s][r] += v * wv;
      }
    }
  }
#pragma unroll
  for (int s = 0; s < 2; ++s) {
#pragma unroll
    for (int r = 0; r < 8; ++r) {
      float v = rs[s][r];
      v += __shfl_xor(v, 1, 32);
      v += __shfl_xor(v, 2, 32);
      v += __shfl_xor(v, 4, 32);
      v += __shfl_xor(v, 8, 32);
      rs[s][r] = v;
    }
  }
  if (l16 == 0) {
    const int slice = blockIdx.y * 4 + wn;  // 0..15, each covers 64 columns
#pragma unroll
    for (int s = 0; s < 2; ++s) {
#pragma unroll
      for (int r = 0; r < 8; ++r) {
        const int m = m0 + s * 16 + hf * 8 + r;
        if (m < M) part[(size_t)slice * partStride + m] = rs[s][r];
      }
    }
  }
}

// ---------------------------------------------------------------------------
// GEMM 3: pairwise layer 1. A rows p=(m,a) built on the fly from mention table.
// ---------------------------------------------------------------------------
__global__ __launch_bounds__(256) void gemm_pair_relu_h16(
    const _Float16* __restrict__ me, int R,
    const _Float16* __restrict__ BT, const float* __restrict__ bias,
    _Float16* __restrict__ C) {
  const int Kd = 2304, N = 1024;
  const int wid = threadIdx.x >> 5, lane = threadIdx.x & 31;
  const int hf = lane >> 4, l16 = lane & 15;
  const int wm = wid & 1, wn = wid >> 1;
  const int m0 = blockIdx.x * 64 + wm * 32;
  const int n0 = blockIdx.y * 256 + wn * 64;
  int p0 = m0 + l16;      if (p0 >= R) p0 = R - 1;
  int p1 = m0 + 16 + l16; if (p1 >= R) p1 = R - 1;
  const int mA = p0 / 50, aA = p0 % 50;
  const int mB = p1 / 50, aB = p1 % 50;
  const int jrA = mA - aA - 1, jrB = mB - aB - 1;
  const int jA = jrA < 0 ? 0 : jrA;
  const int jB = jrB < 0 ? 0 : jrB;

  v8f acc[2][4] = {};
  for (int k0 = 0; k0 < Kd; k0 += 32) {
    const int ka = k0 + hf * 8;
    v16h af0, af1;
    ((v8h*)&af0)[0] = pair_chunk(me, mA, jA, ka);
    ((v8h*)&af0)[1] = pair_chunk(me, mA, jA, ka + 16);
    ((v8h*)&af1)[0] = pair_chunk(me, mB, jB, ka);
    ((v8h*)&af1)[1] = pair_chunk(me, mB, jB, ka + 16);
#pragma unroll
    for (int t = 0; t < 4; ++t) {
      v16h bf = load_bfrag(BT, (size_t)(n0 + t * 16 + l16), Kd, k0, hf);
      acc[0][t] = WMMA16(af0, bf, acc[0][t]);
      acc[1][t] = WMMA16(af1, bf, acc[1][t]);
    }
  }
#pragma unroll
  for (int s = 0; s < 2; ++s) {
#pragma unroll
    for (int t = 0; t < 4; ++t) {
      const int n = n0 + t * 16 + l16;
      const float bv = bias[n];
#pragma unroll
      for (int r = 0; r < 8; ++r) {
        const int m = m0 + s * 16 + hf * 8 + r;
        if (m < R) {
          float v = acc[s][t][r] + bv;
          v = v > 0.f ? v : 0.f;
          __builtin_nontemporal_store((_Float16)v, &C[(size_t)m * N + n]);
        }
      }
    }
  }
}

// ---------------------------------------------------------------------------
// Small kernels
// ---------------------------------------------------------------------------
__global__ void reduce16(const float* __restrict__ part, int partStride, int M,
                         const float* __restrict__ b2, float* __restrict__ outv) {
  const int m = blockIdx.x * blockDim.x + threadIdx.x;
  if (m < M) {
    float s = b2[0];
#pragma unroll
    for (int q = 0; q < 16; ++q) s += part[(size_t)q * partStride + m];
    outv[m] = s;
  }
}

__global__ void transpose_f32_to_f16(const float* __restrict__ W, _Float16* __restrict__ WT,
                                     int Kd, int N) {
  const int i = blockIdx.x * blockDim.x + threadIdx.x;
  if (i >= Kd * N) return;
  const int k = i / N, n = i % N;
  WT[(size_t)n * Kd + k] = (_Float16)W[i];
}

__global__ void init_hist(unsigned* hist, unsigned* state) {
  const int i = blockIdx.x * blockDim.x + threadIdx.x;
  if (i < 1024) hist[i] = 0;
  if (i < 8) state[i] = 0;
}

__global__ void make_keys(const float* __restrict__ scores, unsigned* __restrict__ keys, int n) {
  const int i = blockIdx.x * blockDim.x + threadIdx.x;
  if (i < n) {
    const unsigned u = __float_as_uint(scores[i]);
    keys[i] = (u & 0x80000000u) ? ~u : (u | 0x80000000u);
  }
}

__global__ void topk_hist(const unsigned* __restrict__ keys, int n, unsigned* __restrict__ hist,
                          const unsigned* __restrict__ state, int pass) {
  const unsigned pfx = state[0];
  const int stride = gridDim.x * blockDim.x;
  for (int i = blockIdx.x * blockDim.x + threadIdx.x; i < n; i += stride) {
    const unsigned k = keys[i];
    if (pass == 0 || (k >> (32 - 8 * pass)) == pfx)
      atomicAdd(&hist[pass * 256 + ((k >> (24 - 8 * pass)) & 255u)], 1u);
  }
}

__global__ void topk_select(const unsigned* __restrict__ hist, unsigned* __restrict__ state,
                            int pass, int Ksel) {
  if (threadIdx.x != 0 || blockIdx.x != 0) return;
  unsigned rem = (pass == 0) ? (unsigned)Ksel : state[1];
  unsigned acc = 0;
  for (int b = 255; b >= 0; --b) {
    const unsigned c = hist[pass * 256 + b];
    if (acc + c >= rem) {
      state[0] = (pass == 0) ? (unsigned)b : ((state[0] << 8) | (unsigned)b);
      state[1] = rem - acc;
      return;
    }
    acc += c;
  }
}

__global__ void topk_blockcount(const unsigned* __restrict__ keys, int n,
                                const unsigned* __restrict__ state,
                                int* __restrict__ gcnt, int* __restrict__ ecnt) {
  const unsigned T = state[0];
  const int i = blockIdx.x * 256 + threadIdx.x;
  const bool g = (i < n) && (keys[i] > T);
  const bool e = (i < n) && (keys[i] == T);
  const unsigned gm = (unsigned)__ballot(g);
  const unsigned em = (unsigned)__ballot(e);
  __shared__ int sg[8], se[8];
  const int wid = threadIdx.x >> 5, lane = threadIdx.x & 31;
  if (lane == 0) { sg[wid] = __popc(gm); se[wid] = __popc(em); }
  __syncthreads();
  if (threadIdx.x == 0) {
    int a = 0, b = 0;
    for (int w = 0; w < 8; ++w) { a += sg[w]; b += se[w]; }
    gcnt[blockIdx.x] = a; ecnt[blockIdx.x] = b;
  }
}

__global__ void topk_scanblocks(const int* __restrict__ gcnt, const int* __restrict__ ecnt,
                                int* __restrict__ goff, int* __restrict__ eoff, int nblk) {
  if (threadIdx.x != 0 || blockIdx.x != 0) return;
  int ga = 0, ea = 0;
  for (int b = 0; b < nblk; ++b) {
    goff[b] = ga; eoff[b] = ea;
    ga += gcnt[b]; ea += ecnt[b];
  }
}

__global__ void topk_scatter(const unsigned* __restrict__ keys, int n,
                             const unsigned* __restrict__ state,
                             const int* __restrict__ goff, const int* __restrict__ eoff,
                             int* __restrict__ top_idx) {
  const unsigned T = state[0];
  const int r = (int)state[1];
  const int i = blockIdx.x * 256 + threadIdx.x;
  const bool g = (i < n) && (keys[i] > T);
  const bool e = (i < n) && (keys[i] == T);
  const unsigned gm = (unsigned)__ballot(g);
  const unsigned em = (unsigned)__ballot(e);
  const int wid = threadIdx.x >> 5, lane = threadIdx.x & 31;
  __shared__ int sg[8], se[8];
  if (lane == 0) { sg[wid] = __popc(gm); se[wid] = __popc(em); }
  __syncthreads();
  int gw = 0, ew = 0;
  for (int w = 0; w < wid; ++w) { gw += sg[w]; ew += se[w]; }
  const unsigned lt = (1u << lane) - 1u;
  const int gex = goff[blockIdx.x] + gw + __popc(gm & lt);
  const int eex = eoff[blockIdx.x] + ew + __popc(em & lt);
  if (g || (e && eex < r)) {
    const int emin = eex < r ? eex : r;
    top_idx[gex + emin] = i;  // ascending-index output by construction
  }
}

__global__ void gather_kernel(const float* __restrict__ emb, const float* __restrict__ scores,
                              const int* __restrict__ doc, const int* __restrict__ top_idx,
                              _Float16* __restrict__ me, float* __restrict__ ms,
                              int* __restrict__ mdoc, int K, int E) {
  const int t = blockIdx.x * blockDim.x + threadIdx.x;
  if (t < K) {
    const int idx = top_idx[t];
    ms[t] = scores[idx];
    mdoc[t] = doc[idx];
  }
  if (t < K * E) {
    const int i = t / E, k = t % E;
    me[t] = (_Float16)emb[(size_t)top_idx[i] * E + k];
  }
}

__global__ void out_kernel(const float* __restrict__ ps, const float* __restrict__ ms,
                           const int* __restrict__ mdoc, float* __restrict__ out, int K, int A) {
  const int t = blockIdx.x * blockDim.x + threadIdx.x;
  if (t >= K * (A + 1)) return;
  const int m = t / (A + 1), c = t % (A + 1);
  if (c == 0) { out[t] = 0.f; return; }
  const int a = c - 1;
  const int jr = m - a - 1;
  const int j = jr < 0 ? 0 : jr;
  const bool ok = (jr >= 0) && (mdoc[m] == mdoc[j]);
  const float v = ps[m * A + a] + ms[m] + ms[j];
  out[t] = ok ? v : -INFINITY;
}

// ---------------------------------------------------------------------------
extern "C" void kernel_launch(void* const* d_in, const int* in_sizes, int n_in,
                              void* d_out, int out_size, void* d_ws, size_t ws_size,
                              hipStream_t stream) {
  const float* emb = (const float*)d_in[0];
  const int*   doc = (const int*)d_in[1];
  const float* uw0 = (const float*)d_in[2];  const float* ub0 = (const float*)d_in[3];
  const float* uw1 = (const float*)d_in[4];  const float* ub1 = (const float*)d_in[5];
  const float* uw2 = (const float*)d_in[6];  const float* ub2 = (const float*)d_in[7];
  const float* pw0 = (const float*)d_in[8];  const float* pb0 = (const float*)d_in[9];
  const float* pw1 = (const float*)d_in[10]; const float* pb1 = (const float*)d_in[11];
  const float* pw2 = (const float*)d_in[12]; const float* pb2 = (const float*)d_in[13];

  const int N = 100000, E = 768, H = 1024, K = 1500, A = 50;
  const int R = K * A;                 // 75000 pair rows
  const int nblk = (N + 255) / 256;    // 391

  char* ws = (char*)d_ws;
  size_t off = 0;
  auto alloc = [&](size_t bytes) -> char* {
    char* p = ws + off;
    off = (off + bytes + 255) & ~(size_t)255;
    return p;
  };
  unsigned*  keys    = (unsigned*)alloc((size_t)N * 4);
  unsigned*  hist    = (unsigned*)alloc(1024 * 4);
  unsigned*  state   = (unsigned*)alloc(64);
  int*       gcnt    = (int*)alloc((size_t)nblk * 4);
  int*       ecnt    = (int*)alloc((size_t)nblk * 4);
  int*       goff    = (int*)alloc((size_t)nblk * 4);
  int*       eoff    = (int*)alloc((size_t)nblk * 4);
  int*       top_idx = (int*)alloc((size_t)K * 4);
  float*     ms      = (float*)alloc((size_t)K * 4);
  int*       mdoc    = (int*)alloc((size_t)K * 4);
  float*     scores  = (float*)alloc((size_t)N * 4);
  float*     pscores = (float*)alloc((size_t)R * 4);
  float*     part    = (float*)alloc((size_t)16 * N * 4);     // fused-dot partials
  _Float16*  meh     = (_Float16*)alloc((size_t)K * E * 2);
  _Float16*  uw0T    = (_Float16*)alloc((size_t)H * E * 2);
  _Float16*  uw1T    = (_Float16*)alloc((size_t)H * H * 2);
  _Float16*  pw0T    = (_Float16*)alloc((size_t)H * 3 * E * 2);
  _Float16*  pw1T    = (_Float16*)alloc((size_t)H * H * 2);
  _Float16*  h1      = (_Float16*)alloc((size_t)N * H * 2);   // reused as pairwise hidden

  const dim3 b256(256);

  // weight prep + state init
  init_hist<<<4, b256, 0, stream>>>(hist, state);
  transpose_f32_to_f16<<<(E * H + 255) / 256, b256, 0, stream>>>(uw0, uw0T, E, H);
  transpose_f32_to_f16<<<(H * H + 255) / 256, b256, 0, stream>>>(uw1, uw1T, H, H);
  transpose_f32_to_f16<<<(3 * E * H + 255) / 256, b256, 0, stream>>>(pw0, pw0T, 3 * E, H);
  transpose_f32_to_f16<<<(H * H + 255) / 256, b256, 0, stream>>>(pw1, pw1T, H, H);

  // unary FFNN (block tile 64 x 256, gridDim.y = 1024/256 = 4)
  const dim3 g1((N + 63) / 64, H / 256);
  gemm_a32_relu_h16<<<g1, b256, 0, stream>>>(emb, N, E, uw0T, ub0, h1, H);
  gemm_a16_relu_dot<<<g1, b256, 0, stream>>>(h1, N, H, uw1T, ub1, uw2, part, N, H);
  reduce16<<<(N + 255) / 256, b256, 0, stream>>>(part, N, N, ub2, scores);

  // exact top-K with sorted-ascending indices
  make_keys<<<nblk, b256, 0, stream>>>(scores, keys, N);
  for (int p = 0; p < 4; ++p) {
    topk_hist<<<nblk, b256, 0, stream>>>(keys, N, hist, state, p);
    topk_select<<<1, 32, 0, stream>>>(hist, state, p, K);
  }
  topk_blockcount<<<nblk, b256, 0, stream>>>(keys, N, state, gcnt, ecnt);
  topk_scanblocks<<<1, 32, 0, stream>>>(gcnt, ecnt, goff, eoff, nblk);
  topk_scatter<<<nblk, b256, 0, stream>>>(keys, N, state, goff, eoff, top_idx);
  gather_kernel<<<(K * E + 255) / 256, b256, 0, stream>>>(emb, scores, doc, top_idx,
                                                          meh, ms, mdoc, K, E);

  // pairwise FFNN
  const dim3 g2((R + 63) / 64, H / 256);
  gemm_pair_relu_h16<<<g2, b256, 0, stream>>>(meh, R, pw0T, pb0, h1);
  gemm_a16_relu_dot<<<g2, b256, 0, stream>>>(h1, R, H, pw1T, pb1, pw2, part, N, H);
  reduce16<<<(R + 255) / 256, b256, 0, stream>>>(part, N, R, pb2, pscores);

  // final assembly [K, A+1]
  out_kernel<<<(K * (A + 1) + 255) / 256, b256, 0, stream>>>(pscores, ms, mdoc,
                                                             (float*)d_out, K, A);
  (void)in_sizes; (void)n_in; (void)out_size; (void)ws_size;
}
